// DepthLSSTransform_42107859370544
// MI455X (gfx1250) — compile-verified, matrix operand order checked by reading, not run on hardware
//
#include <hip/hip_runtime.h>
#include <math.h>

// ---------------- problem constants ----------------
#define BN_CAMS   6
#define CIN       256
#define FH        32
#define FW        88
#define IH        256
#define IW        704
#define DBINS     59
#define CCH       80
#define NXG       360
#define NYG       360
#define FHW       (FH * FW)                 // 2816
#define NPTS      (BN_CAMS * DBINS * FHW)   // 996864
#define BEVHW     (NXG * NYG)               // 129600

typedef __bf16 bf16_t;
typedef __attribute__((ext_vector_type(16))) bf16_t v16bf;
typedef __attribute__((ext_vector_type(8)))  bf16_t v8bf;
typedef __attribute__((ext_vector_type(8)))  float  v8f;

// ---------------- generic implicit-GEMM conv via WMMA bf16 ----------------
// out[co,p] = sum_{r,s,ci} W[co,ci,r,s] * in[n,ci,oh*st-pad+r,ow*st-pad+s]
// Block tile: 32 Cout x 128 pixels, 4 waves; each wave owns 32 pixels and
// computes a 2x2 grid of 16x16 WMMA accumulators (4 v_wmma per K-step).
// K-loop runs (r, s, ci0+=32): no integer division in the hot loop.
// LDS tiles are K-contiguous with 80B row stride: all fragment reads are
// 16B-aligned, bank-conflict-free ds_load_b128; staging uses ds_store_b128.
struct ConvParams {
    const float* in;      // NCHW f32
    const float* wgt;     // OIHW f32
    const float* bias;    // nullable
    const float* gamma;   // nullable (BN scale)
    const float* beta;    // BN shift
    float* out;
    int Nimg, Cin, H, W, Cout, KH, KW, stride, pad, OH, OW;
    int outCtotal, outCoffset;
    int relu;
};

#define BM 32          // Cout per block (2 M-tiles)
#define BNP 128        // pixels per block (4 waves x 2 N-tiles)
#define BK 32          // K-step
#define BKP 40         // padded row stride: 80B, 16B-aligned, conflict-free

__device__ __forceinline__ v16bf ldsfrag(const bf16_t* row, int hi) {
    // Documented 16-bit wave32 fragment: lane half hi selects K-chunks
    // {hi*8..hi*8+7} and {16+hi*8..16+hi*8+7} -> two b128 LDS loads.
    v8bf lo = *(const v8bf*)(row + hi * 8);
    v8bf up = *(const v8bf*)(row + 16 + hi * 8);
    return __builtin_shufflevector(lo, up, 0, 1, 2, 3, 4, 5, 6, 7,
                                   8, 9, 10, 11, 12, 13, 14, 15);
}

__global__ __launch_bounds__(128)
void conv_wmma_kernel(ConvParams p) {
    __shared__ __align__(16) bf16_t As[BM][BKP];
    __shared__ __align__(16) bf16_t Bs[BNP][BKP];

    const int tid  = threadIdx.x;
    const int lane = tid & 31;
    const int wave = tid >> 5;
    const int coBase  = blockIdx.y * BM;
    const int pixBase = blockIdx.x * BNP;
    const int P   = p.Nimg * p.OH * p.OW;
    const int K   = p.Cin * p.KH * p.KW;
    const int KHW = p.KH * p.KW;
    const int HW  = p.H * p.W;

    v8f acc[2][2] = {};

    // per-thread B column: pixel prow = pixBase + tid (decompose once)
    const int prow = pixBase + tid;
    int bn = 0, boh = 0, bow = 0;
    const bool pvalid = (prow < P);
    if (pvalid) {
        bn = prow / (p.OH * p.OW);
        int r0 = prow - bn * p.OH * p.OW;
        boh = r0 / p.OW;
        bow = r0 - boh * p.OW;
    }
    const int ihBase = boh * p.stride - p.pad;
    const int iwBase = bow * p.stride - p.pad;
    const float* inN = p.in + (size_t)bn * p.Cin * HW;

    // A staging role: thread -> (row m, 8-wide K chunk t0), one b128 store
    const int am  = tid >> 2;          // 0..31
    const int at0 = (tid & 3) * 8;     // 0,8,16,24
    const int aco = coBase + am;

    for (int r = 0; r < p.KH; ++r) {
        const int ih = ihBase + r;
        for (int s = 0; s < p.KW; ++s) {
            const int iw = iwBase + s;
            const bool inb = pvalid && ih >= 0 && ih < p.H && iw >= 0 && iw < p.W;
            const float* src = inN + (size_t)ih * p.W + iw;   // + ci*HW
            const int rs = r * p.KW + s;
            for (int ci0 = 0; ci0 < p.Cin; ci0 += BK) {
                // --- stage A (weights) 32x32: one packed b128 store/thread ---
                {
                    v8bf aw;
                    #pragma unroll
                    for (int t = 0; t < 8; ++t) {
                        int ci = ci0 + at0 + t;
                        float v = (aco < p.Cout && ci < p.Cin)
                                  ? p.wgt[(size_t)aco * K + (size_t)ci * KHW + rs] : 0.f;
                        aw[t] = (bf16_t)v;
                    }
                    *(v8bf*)&As[am][at0] = aw;
                }
                // --- stage B (im2col) 128x32: K-contiguous, b128 stores ---
                #pragma unroll
                for (int t0 = 0; t0 < BK; t0 += 8) {
                    v8bf bw;
                    #pragma unroll
                    for (int t = 0; t < 8; ++t) {
                        int ci = ci0 + t0 + t;
                        float v = (inb && ci < p.Cin) ? src[(size_t)ci * HW] : 0.f;
                        bw[t] = (bf16_t)v;
                    }
                    *(v8bf*)&Bs[tid][t0] = bw;
                }
                __syncthreads();

                // --- fragments via b128 LDS loads + 2x2 MMA ---
                const int mrow = lane & 15;
                const int hi   = lane >> 4;
                v16bf af[2], bfv[2];
                #pragma unroll
                for (int m0 = 0; m0 < 2; ++m0)
                    af[m0] = ldsfrag(&As[m0 * 16 + mrow][0], hi);
                #pragma unroll
                for (int n0 = 0; n0 < 2; ++n0)
                    bfv[n0] = ldsfrag(&Bs[wave * 32 + n0 * 16 + mrow][0], hi);
                #pragma unroll
                for (int m0 = 0; m0 < 2; ++m0)
                    #pragma unroll
                    for (int n0 = 0; n0 < 2; ++n0)
                        acc[m0][n0] = __builtin_amdgcn_wmma_f32_16x16x32_bf16(
                            false, af[m0], false, bfv[n0],
                            (short)0, acc[m0][n0], false, false);
                __syncthreads();
            }
        }
    }

    // --- epilogue: C layout lane l holds N=l%16, VGPR r holds M = r + 8*(l/16) ---
    const int nn = lane & 15;
    #pragma unroll
    for (int n0 = 0; n0 < 2; ++n0) {
        const int prow2 = pixBase + wave * 32 + n0 * 16 + nn;
        if (prow2 < P) {
            int n  = prow2 / (p.OH * p.OW);
            int r0 = prow2 - n * p.OH * p.OW;
            int oh = r0 / p.OW;
            int ow = r0 - oh * p.OW;
            #pragma unroll
            for (int m0 = 0; m0 < 2; ++m0)
                #pragma unroll
                for (int r = 0; r < 8; ++r) {
                    int co = coBase + m0 * 16 + r + 8 * (lane >> 4);
                    if (co < p.Cout) {
                        float v = acc[m0][n0][r];
                        if (p.bias)  v += p.bias[co];
                        if (p.gamma) v = v * p.gamma[co] + p.beta[co];
                        if (p.relu)  v = fmaxf(v, 0.f);
                        p.out[((size_t)(n * p.outCtotal + p.outCoffset + co) * p.OH + oh) * p.OW + ow] = v;
                    }
                }
        }
    }
}

// ---------------- dtransform layer 1: 1x1 conv 1->8 + BN + ReLU ----------------
__global__ void dt1_kernel(const float* __restrict__ d, const float* __restrict__ w,
                           const float* __restrict__ b, const float* __restrict__ g,
                           const float* __restrict__ be, float* __restrict__ out) {
    int i = blockIdx.x * blockDim.x + threadIdx.x;
    const int tot = BN_CAMS * IH * IW;
    if (i >= tot) return;
    int n  = i / (IH * IW);
    int hw = i - n * IH * IW;
    float v = d[i];
    #pragma unroll
    for (int c = 0; c < 8; ++c) {
        float y = w[c] * v + b[c];
        y = y * g[c] + be[c];
        out[(n * 8 + c) * (IH * IW) + hw] = fmaxf(y, 0.f);
    }
}

// ---------------- copy x into concat buffer channels [64,320) ----------------
__global__ void copyx_kernel(const float* __restrict__ x, float* __restrict__ cat) {
    int i = blockIdx.x * blockDim.x + threadIdx.x;
    const int tot = BN_CAMS * CIN * FHW;
    if (i >= tot) return;
    int n   = i / (CIN * FHW);
    int rem = i - n * CIN * FHW;
    int c   = rem / FHW;
    int hw  = rem - c * FHW;
    cat[(n * 320 + 64 + c) * FHW + hw] = x[i];
}

// ---------------- softmax over D=59 channels (in place on dn3) ----------------
__global__ void softmax_kernel(float* __restrict__ dn3) {
    int i = blockIdx.x * blockDim.x + threadIdx.x;
    const int tot = BN_CAMS * FHW;
    if (i >= tot) return;
    int n  = i / FHW;
    int hw = i - n * FHW;
    float* base = dn3 + (size_t)n * 139 * FHW + hw;
    float mx = -1e30f;
    for (int dd = 0; dd < DBINS; ++dd) mx = fmaxf(mx, base[dd * FHW]);
    float s = 0.f;
    for (int dd = 0; dd < DBINS; ++dd) s += __expf(base[dd * FHW] - mx);
    float inv = 1.f / s;
    for (int dd = 0; dd < DBINS; ++dd)
        base[dd * FHW] = __expf(base[dd * FHW] - mx) * inv;
}

// ---------------- per-camera transform precompute ----------------
__device__ __forceinline__ void inv3(const float* a, float* o) {
    float c00 =  a[4]*a[8] - a[5]*a[7];
    float c01 = -(a[3]*a[8] - a[5]*a[6]);
    float c02 =  a[3]*a[7] - a[4]*a[6];
    float det = a[0]*c00 + a[1]*c01 + a[2]*c02;
    float id  = 1.f / det;
    o[0] = c00 * id;
    o[1] = -(a[1]*a[8] - a[2]*a[7]) * id;
    o[2] =  (a[1]*a[5] - a[2]*a[4]) * id;
    o[3] = c01 * id;
    o[4] =  (a[0]*a[8] - a[2]*a[6]) * id;
    o[5] = -(a[0]*a[5] - a[2]*a[3]) * id;
    o[6] = c02 * id;
    o[7] = -(a[0]*a[7] - a[1]*a[6]) * id;
    o[8] =  (a[0]*a[4] - a[1]*a[3]) * id;
}

__global__ void xform_kernel(const float* __restrict__ c2l_rot,
                             const float* __restrict__ c2l_tr,
                             const float* __restrict__ intrins,
                             const float* __restrict__ post_rots,
                             const float* __restrict__ post_trans,
                             float* __restrict__ xf) {
    int n = threadIdx.x;
    if (n >= BN_CAMS) return;
    float pri[9], Ki[9], comb[9];
    inv3(post_rots + n * 9, pri);
    inv3(intrins + n * 9, Ki);
    const float* R = c2l_rot + n * 9;
    #pragma unroll
    for (int r = 0; r < 3; ++r)
        #pragma unroll
        for (int c = 0; c < 3; ++c)
            comb[r * 3 + c] = R[r*3+0]*Ki[0*3+c] + R[r*3+1]*Ki[1*3+c] + R[r*3+2]*Ki[2*3+c];
    float* o = xf + n * 24;
    #pragma unroll
    for (int k = 0; k < 9; ++k) { o[k] = pri[k]; o[9 + k] = comb[k]; }
    #pragma unroll
    for (int k = 0; k < 3; ++k) { o[18 + k] = post_trans[n*3 + k]; o[21 + k] = c2l_tr[n*3 + k]; }
}

// ---------------- voxelize frustum points ----------------
__global__ void point_kernel(const float* __restrict__ xf, int* __restrict__ pidx) {
    int i = blockIdx.x * blockDim.x + threadIdx.x;
    if (i >= NPTS) return;
    int n   = i / (DBINS * FHW);
    int rem = i - n * DBINS * FHW;
    int dd  = rem / FHW;
    int hw  = rem - dd * FHW;
    int h   = hw / FW;
    int w   = hw - h * FW;
    float xs = (float)w * ((float)(IW - 1) / (float)(FW - 1));
    float ys = (float)h * ((float)(IH - 1) / (float)(FH - 1));
    float ds = (float)dd + 1.f;
    const float* o = xf + n * 24;
    float px = xs - o[18], py = ys - o[19], pz = ds - o[20];
    float qx = o[0]*px + o[1]*py + o[2]*pz;
    float qy = o[3]*px + o[4]*py + o[5]*pz;
    float qz = o[6]*px + o[7]*py + o[8]*pz;
    qx *= qz; qy *= qz;
    float rx = o[ 9]*qx + o[10]*qy + o[11]*qz + o[21];
    float ry = o[12]*qx + o[13]*qy + o[14]*qz + o[22];
    float rz = o[15]*qx + o[16]*qy + o[17]*qz + o[23];
    int xc = (int)((rx + 54.f) / 0.3f);   // trunc toward zero, like astype(int32)
    int yc = (int)((ry + 54.f) / 0.3f);
    int zc = (int)((rz + 10.f) / 20.f);
    int idx = -1;
    if (xc >= 0 && xc < NXG && yc >= 0 && yc < NYG && zc == 0)
        idx = xc * NYG + yc;
    pidx[i] = idx;
}

// ---------------- scatter-sum lift into BEV (atomics resolve in L2) ----------------
__global__ void scatter_kernel(const int* __restrict__ pidx,
                               const float* __restrict__ dn3,
                               float* __restrict__ bev) {
    int p = blockIdx.x * blockDim.x + threadIdx.x;
    int c = blockIdx.y;
    if (p >= NPTS) return;
    int idx = pidx[p];
    if (idx < 0) return;
    int n   = p / (DBINS * FHW);
    int rem = p - n * DBINS * FHW;
    int dd  = rem / FHW;
    int hw  = rem - dd * FHW;
    const float* base = dn3 + (size_t)n * 139 * FHW + hw;
    float depth = base[dd * FHW];
    float feat  = base[(DBINS + c) * FHW];
    atomicAdd(bev + (size_t)c * BEVHW + idx, depth * feat);
}

__global__ void zero_kernel(float* __restrict__ buf, int tot) {
    int i = blockIdx.x * blockDim.x + threadIdx.x;
    for (; i < tot; i += gridDim.x * blockDim.x) buf[i] = 0.f;
}

// ---------------- host-side launch ----------------
extern "C" void kernel_launch(void* const* d_in, const int* in_sizes, int n_in,
                              void* d_out, int out_size, void* d_ws, size_t ws_size,
                              hipStream_t stream) {
    (void)in_sizes; (void)n_in; (void)out_size; (void)ws_size;
    const float* x_feat     = (const float*)d_in[0];
    const float* d_depth    = (const float*)d_in[1];
    const float* c2l_rot    = (const float*)d_in[2];
    const float* c2l_tr     = (const float*)d_in[3];
    const float* intrins    = (const float*)d_in[4];
    const float* post_rots  = (const float*)d_in[5];
    const float* post_trans = (const float*)d_in[6];
    #define PRM(k) ((const float*)d_in[7 + (k)])
    // 0..3 dt1(w,b,g,be) ; 4..7 dt2 ; 8..11 dt3 ;
    // 12..15 dn1 ; 16..19 dn2 ; 20..21 dn_w3,dn_b3 ;
    // 22..24 ds1(w,g,be) ; 25..27 ds2 ; 28..30 ds3

    float* ws = (float*)d_ws;
    const size_t F_H1   = 0;                                   // 6*8*256*704
    const size_t F_H2   = F_H1  + (size_t)BN_CAMS*8*IH*IW;     // 6*32*64*176
    const size_t F_CAT  = F_H2  + (size_t)BN_CAMS*32*64*176;   // 6*320*32*88
    const size_t F_DN1  = F_CAT + (size_t)BN_CAMS*320*FHW;
    const size_t F_DN2  = F_DN1 + (size_t)BN_CAMS*256*FHW;
    const size_t F_DN3  = F_DN2 + (size_t)BN_CAMS*256*FHW;
    const size_t F_XF   = F_DN3 + (size_t)BN_CAMS*139*FHW;
    const size_t F_PIDX = F_XF  + (size_t)BN_CAMS*24;
    const size_t F_BEV  = F_PIDX + (size_t)NPTS;
    const size_t F_Y1   = F_H1;    // reuse: dtransform buffers dead by now
    const size_t F_Y2   = F_CAT;   // reuse

    auto conv = [&](const float* in, const float* w, const float* b,
                    const float* g, const float* be, float* out,
                    int Nimg, int Cin, int H, int W, int Cout,
                    int KH, int KW, int st, int pd, int OH, int OW,
                    int outCtot, int outCoff, int relu) {
        ConvParams p;
        p.in = in; p.wgt = w; p.bias = b; p.gamma = g; p.beta = be; p.out = out;
        p.Nimg = Nimg; p.Cin = Cin; p.H = H; p.W = W; p.Cout = Cout;
        p.KH = KH; p.KW = KW; p.stride = st; p.pad = pd; p.OH = OH; p.OW = OW;
        p.outCtotal = outCtot; p.outCoffset = outCoff; p.relu = relu;
        int P = Nimg * OH * OW;
        dim3 grid((P + BNP - 1) / BNP, (Cout + BM - 1) / BM);
        hipLaunchKernelGGL(conv_wmma_kernel, grid, dim3(128), 0, stream, p);
    };

    // ---- dtransform ----
    {
        int tot = BN_CAMS * IH * IW;
        hipLaunchKernelGGL(dt1_kernel, dim3((tot + 255) / 256), dim3(256), 0, stream,
                           d_depth, PRM(0), PRM(1), PRM(2), PRM(3), ws + F_H1);
    }
    conv(ws + F_H1, PRM(4), PRM(5), PRM(6), PRM(7), ws + F_H2,
         BN_CAMS, 8, IH, IW, 32, 5, 5, 4, 2, 64, 176, 32, 0, 1);
    conv(ws + F_H2, PRM(8), PRM(9), PRM(10), PRM(11), ws + F_CAT,
         BN_CAMS, 32, 64, 176, 64, 5, 5, 2, 2, FH, FW, 320, 0, 1);
    {
        int tot = BN_CAMS * CIN * FHW;
        hipLaunchKernelGGL(copyx_kernel, dim3((tot + 255) / 256), dim3(256), 0, stream,
                           x_feat, ws + F_CAT);
    }
    // ---- depthnet ----
    conv(ws + F_CAT, PRM(12), PRM(13), PRM(14), PRM(15), ws + F_DN1,
         BN_CAMS, 320, FH, FW, 256, 3, 3, 1, 1, FH, FW, 256, 0, 1);
    conv(ws + F_DN1, PRM(16), PRM(17), PRM(18), PRM(19), ws + F_DN2,
         BN_CAMS, 256, FH, FW, 256, 3, 3, 1, 1, FH, FW, 256, 0, 1);
    conv(ws + F_DN2, PRM(20), PRM(21), nullptr, nullptr, ws + F_DN3,
         BN_CAMS, 256, FH, FW, 139, 1, 1, 1, 0, FH, FW, 139, 0, 0);
    {
        int tot = BN_CAMS * FHW;
        hipLaunchKernelGGL(softmax_kernel, dim3((tot + 255) / 256), dim3(256), 0, stream,
                           ws + F_DN3);
    }
    // ---- geometry + bev_pool ----
    hipLaunchKernelGGL(xform_kernel, dim3(1), dim3(8), 0, stream,
                       c2l_rot, c2l_tr, intrins, post_rots, post_trans, ws + F_XF);
    hipLaunchKernelGGL(zero_kernel, dim3(1024), dim3(256), 0, stream,
                       ws + F_BEV, CCH * BEVHW);
    hipLaunchKernelGGL(point_kernel, dim3((NPTS + 255) / 256), dim3(256), 0, stream,
                       ws + F_XF, (int*)(ws + F_PIDX));
    hipLaunchKernelGGL(scatter_kernel, dim3((NPTS + 255) / 256, CCH), dim3(256), 0, stream,
                       (const int*)(ws + F_PIDX), ws + F_DN3, ws + F_BEV);
    // ---- downsample ----
    conv(ws + F_BEV, PRM(22), nullptr, PRM(23), PRM(24), ws + F_Y1,
         1, CCH, NXG, NYG, CCH, 3, 3, 1, 1, NXG, NYG, CCH, 0, 1);
    conv(ws + F_Y1, PRM(25), nullptr, PRM(26), PRM(27), ws + F_Y2,
         1, CCH, NXG, NYG, CCH, 3, 3, 2, 1, 180, 180, CCH, 0, 1);
    conv(ws + F_Y2, PRM(28), nullptr, PRM(29), PRM(30), (float*)d_out,
         1, CCH, 180, 180, CCH, 3, 3, 1, 1, 180, 180, CCH, 0, 1);
    #undef PRM
}
